// ChronotopeBlockV2_84198538870848
// MI455X (gfx1250) — matrix operand order repaired and checked by reading, not compiled, not run on hardware
//
#include <hip/hip_runtime.h>

// ---------------------------------------------------------------------------
// Problem constants (from the reference)
// ---------------------------------------------------------------------------
#define B_DIM 2
#define T_DIM 2048
#define C_DIM 1024
#define H_DIM 16
#define HD    64
#define M_MEM 16
#define BT    (B_DIM * T_DIM)   // 4096

typedef __bf16 bf16_t;
typedef bf16_t v16bf __attribute__((ext_vector_type(16)));
typedef bf16_t v8bf  __attribute__((ext_vector_type(8)));
typedef float  v8f   __attribute__((ext_vector_type(8)));

// ---------------------------------------------------------------------------
// bf16 <-> f32 helpers (round-to-nearest-even)
// ---------------------------------------------------------------------------
__device__ __forceinline__ bf16_t f2bf(float f) {
    union { float f; unsigned u; } v; v.f = f;
    unsigned r = v.u + 0x7FFFu + ((v.u >> 16) & 1u);
    unsigned short h = (unsigned short)(r >> 16);
    return __builtin_bit_cast(bf16_t, h);
}
__device__ __forceinline__ float gelu_exact(float x) {
    return 0.5f * x * (1.0f + erff(x * 0.70710678118654752f));
}

// ---------------------------------------------------------------------------
// WMMA helpers (wave32; A/B fragments use contiguous k = 16*(lane/16) + elem)
// ---------------------------------------------------------------------------
__device__ __forceinline__ v8f wmma_bf16(v16bf a, v16bf b, v8f c) {
    return __builtin_amdgcn_wmma_f32_16x16x32_bf16(false, a, false, b,
                                                   (short)0, c, false, false);
}

__device__ __forceinline__ v16bf pack16(v8bf lo, v8bf hi) {
    v16bf r;
#pragma unroll
    for (int j = 0; j < 8; ++j) { r[j] = lo[j]; r[8 + j] = hi[j]; }
    return r;
}

// A fragment: 16x32 bf16 tile, row-major source. Aptr -> element (m0, k0).
// lane<16: row=lane, elems 0..15 = k0..15; lane>=16: k+16.
__device__ __forceinline__ v16bf load_a_frag(const bf16_t* Aptr, long lda, int lane) {
    const bf16_t* p = Aptr + (long)(lane & 15) * lda + ((lane >> 4) << 4);
    return pack16(*(const v8bf*)p, *(const v8bf*)(p + 8));
}

// B fragment from TRANSPOSED source (N x K row-major): element j <-> K row
// ((lane>=16)*16 + j), col = lane&15.  16 contiguous bf16 per lane.
__device__ __forceinline__ v16bf load_bT_frag(const bf16_t* Wt, long ldk, int lane) {
    const bf16_t* p = Wt + (long)(lane & 15) * ldk + ((lane >> 4) << 4);
    return pack16(*(const v8bf*)p, *(const v8bf*)(p + 8));
}

// ---------------------------------------------------------------------------
// Generic bf16 WMMA GEMM:  out = epilogue( A(MxK) @ B(KxN) + bias )
// B supplied TRANSPOSED (Wt: N x K). Block = 256(M) x 64(N): 8 waves, each
// owning a 32x64 tile (2 A-frags, 8 accumulators, 8 WMMA per k-step), share a
// 64x32 Wt tile double-buffered in LDS via global_load_async_to_lds_b128.
// Pipeline per k-step: wait oldest stage -> barrier -> issue next stage ->
// compute (copy of k+1 overlaps WMMA of k).
// act: 0=none 1=gelu 2=tanh
// outB!=null: store bf16 act(acc+bias); else outF = res? + scale*act(acc+bias)
// ---------------------------------------------------------------------------
__global__ __launch_bounds__(256) void gemm_bf16_kernel(
    const bf16_t* __restrict__ A, int lda,
    const bf16_t* __restrict__ Wt, int ldk,
    const float* __restrict__ bias,
    const float* __restrict__ res,
    float* __restrict__ outF,
    bf16_t* __restrict__ outB,
    int M, int N, int K, int act, float scale)
{
    __shared__ __align__(16) bf16_t ldsB[2][64 * 32];   // 2 x 4KB

    int nBlocksN = N >> 6;
    int blockM = blockIdx.x / nBlocksN;
    int ng     = blockIdx.x % nBlocksN;
    int wv     = threadIdx.x >> 5;
    int lane   = threadIdx.x & 31;
    int m0 = (blockM * 8 + wv) * 32;    // 32 M-rows per wave
    int n0 = ng * 64;

    // async-staging assignment: each thread copies one 16B chunk per k-step
    int srow   = threadIdx.x >> 2;      // 0..63
    int schunk = threadIdx.x & 3;       // 0..3 (8 bf16 each)
    unsigned ldsoff0 = (unsigned)(size_t)&ldsB[0][srow * 32 + schunk * 8];
    unsigned ldsoff1 = (unsigned)(size_t)&ldsB[1][srow * 32 + schunk * 8];
    const bf16_t* gB = Wt + (long)(n0 + srow) * ldk + schunk * 8;

    v8f acc[2][4];
#pragma unroll
    for (int mi = 0; mi < 2; ++mi)
#pragma unroll
        for (int ti = 0; ti < 4; ++ti) acc[mi][ti] = v8f{};

    int nK = K >> 5;
    {   // prologue: stage k-step 0 into buffer 0
        unsigned long long ga = (unsigned long long)(size_t)gB;
        asm volatile("global_load_async_to_lds_b128 %0, %1, off"
                     :: "v"(ldsoff0), "v"(ga) : "memory");
    }
    for (int kt = 0; kt < nK; ++kt) {
        // only stage(kt) is outstanding here: drain it
        asm volatile("s_wait_asynccnt 0x0" ::: "memory");
        __syncthreads();   // stage(kt) visible to all; prev compute done reading
        if (kt + 1 < nK) {
            unsigned long long ga =
                (unsigned long long)(size_t)(gB + (long)(kt + 1) * 32);
            unsigned dst = (kt & 1) ? ldsoff0 : ldsoff1;
            asm volatile("global_load_async_to_lds_b128 %0, %1, off"
                         :: "v"(dst), "v"(ga) : "memory");
        }
        int k0 = kt * 32;
        __builtin_prefetch(A + (long)m0 * lda + k0 + 128, 0, 1);
        v16bf a0 = load_a_frag(A + (long)m0 * lda + k0, lda, lane);
        v16bf a1 = load_a_frag(A + (long)(m0 + 16) * lda + k0, lda, lane);
        const bf16_t* curB = ldsB[kt & 1];
#pragma unroll
        for (int ti = 0; ti < 4; ++ti) {
            const bf16_t* p = &curB[(ti * 16 + (lane & 15)) * 32 + ((lane >> 4) << 4)];
            v16bf b = pack16(*(const v8bf*)p, *(const v8bf*)(p + 8));
            acc[0][ti] = wmma_bf16(a0, b, acc[0][ti]);
            acc[1][ti] = wmma_bf16(a1, b, acc[1][ti]);
        }
    }

    int col   = lane & 15;
    int rbase = (lane >> 4) ? 8 : 0;
#pragma unroll
    for (int mi = 0; mi < 2; ++mi) {
#pragma unroll
        for (int ti = 0; ti < 4; ++ti) {
            int n = n0 + ti * 16 + col;
            float bv = bias ? bias[n] : 0.0f;
#pragma unroll
            for (int r = 0; r < 8; ++r) {
                long idx = (long)(m0 + mi * 16 + rbase + r) * N + n;
                float v = acc[mi][ti][r] + bv;
                if (act == 1)      v = gelu_exact(v);
                else if (act == 2) v = tanhf(v);
                if (outB) outB[idx] = f2bf(v);
                else      outF[idx] = (res ? res[idx] : 0.0f) + scale * v;
            }
        }
    }
}

// ---------------------------------------------------------------------------
// Conv-as-GEMM: out += 0.5 * ( Xshift(BTx3C) @ Wcat(3CxC) + conv_b )
// Wt = Wcat^T (C x 3C). A element (row=(b,t), kk=tap*C+ci) = xbf[b,t+tap-1,ci]
// Same 256x64 block / 32x64 wave structure with double-buffered LDS staging.
// ---------------------------------------------------------------------------
__device__ __forceinline__ v16bf load_a_conv(const bf16_t* xbf, int m0, int k0, int lane) {
    int row = m0 + (lane & 15);
    int bb  = row >> 11;            // / T_DIM
    int t   = row & (T_DIM - 1);
    int kb  = k0 + ((lane >> 4) << 4);
    v16bf a;
#pragma unroll
    for (int h = 0; h < 2; ++h) {
        int kk  = kb + h * 8;
        int tap = kk >> 10;         // / C_DIM
        int ci  = kk & (C_DIM - 1);
        int ts  = t + tap - 1;
        if (ts >= 0 && ts < T_DIM) {
            v8bf part = *(const v8bf*)(xbf + ((long)(bb * T_DIM + ts)) * C_DIM + ci);
#pragma unroll
            for (int j = 0; j < 8; ++j) a[h * 8 + j] = part[j];
        } else {
#pragma unroll
            for (int j = 0; j < 8; ++j) a[h * 8 + j] = f2bf(0.0f);
        }
    }
    return a;
}

__global__ __launch_bounds__(256) void conv_gemm_kernel(
    const bf16_t* __restrict__ xbf,
    const bf16_t* __restrict__ Wt,   // (C x 3C): Wt[co][tap*C+ci]
    const float* __restrict__ bias,
    float* __restrict__ outF)        // accumulate 0.5*(acc+bias)
{
    __shared__ __align__(16) bf16_t ldsB[2][64 * 32];

    const int K = 3 * C_DIM, N = C_DIM, ldk = 3 * C_DIM;
    int nBlocksN = N >> 6;
    int blockM = blockIdx.x / nBlocksN;
    int ng     = blockIdx.x % nBlocksN;
    int wv     = threadIdx.x >> 5;
    int lane   = threadIdx.x & 31;
    int m0 = (blockM * 8 + wv) * 32;
    int n0 = ng * 64;

    int srow   = threadIdx.x >> 2;
    int schunk = threadIdx.x & 3;
    unsigned ldsoff0 = (unsigned)(size_t)&ldsB[0][srow * 32 + schunk * 8];
    unsigned ldsoff1 = (unsigned)(size_t)&ldsB[1][srow * 32 + schunk * 8];
    const bf16_t* gB = Wt + (long)(n0 + srow) * ldk + schunk * 8;

    v8f acc[2][4];
#pragma unroll
    for (int mi = 0; mi < 2; ++mi)
#pragma unroll
        for (int ti = 0; ti < 4; ++ti) acc[mi][ti] = v8f{};

    int nK = K >> 5;
    {
        unsigned long long ga = (unsigned long long)(size_t)gB;
        asm volatile("global_load_async_to_lds_b128 %0, %1, off"
                     :: "v"(ldsoff0), "v"(ga) : "memory");
    }
    for (int kt = 0; kt < nK; ++kt) {
        asm volatile("s_wait_asynccnt 0x0" ::: "memory");
        __syncthreads();
        if (kt + 1 < nK) {
            unsigned long long ga =
                (unsigned long long)(size_t)(gB + (long)(kt + 1) * 32);
            unsigned dst = (kt & 1) ? ldsoff0 : ldsoff1;
            asm volatile("global_load_async_to_lds_b128 %0, %1, off"
                         :: "v"(dst), "v"(ga) : "memory");
        }
        int k0 = kt * 32;
        v16bf a0 = load_a_conv(xbf, m0, k0, lane);
        v16bf a1 = load_a_conv(xbf, m0 + 16, k0, lane);
        const bf16_t* curB = ldsB[kt & 1];
#pragma unroll
        for (int ti = 0; ti < 4; ++ti) {
            const bf16_t* p = &curB[(ti * 16 + (lane & 15)) * 32 + ((lane >> 4) << 4)];
            v16bf b = pack16(*(const v8bf*)p, *(const v8bf*)(p + 8));
            acc[0][ti] = wmma_bf16(a0, b, acc[0][ti]);
            acc[1][ti] = wmma_bf16(a1, b, acc[1][ti]);
        }
    }
    int col = lane & 15, rbase = (lane >> 4) ? 8 : 0;
#pragma unroll
    for (int mi = 0; mi < 2; ++mi) {
#pragma unroll
        for (int ti = 0; ti < 4; ++ti) {
            int n = n0 + ti * 16 + col;
            float bv = bias[n];
#pragma unroll
            for (int r = 0; r < 8; ++r) {
                long idx = (long)(m0 + mi * 16 + rbase + r) * C_DIM + n;
                outF[idx] += 0.5f * (acc[mi][ti][r] + bv);
            }
        }
    }
}

// ---------------------------------------------------------------------------
// Flash attention: one wave per (b, h, 16-row q tile); online softmax over
// 32-wide KV chunks; S bounced through LDS; P@V via WMMA using V^T layout.
// qkv: (BT x 3C) bf16. vT: [b][h][d][t] bf16. Output bf16 (BT x C).
// ---------------------------------------------------------------------------
__global__ __launch_bounds__(32) void attn_kernel(
    const bf16_t* __restrict__ qkv, const bf16_t* __restrict__ vT,
    bf16_t* __restrict__ outA)
{
    __shared__ float S[16][36];
    __shared__ float rowm[16], rowl[16], rowc[16];

    int wid = blockIdx.x;                 // B*H*(T/16) total
    int qt  = wid % (T_DIM / 16);
    int tmp = wid / (T_DIM / 16);
    int h   = tmp % H_DIM;
    int b   = tmp / H_DIM;
    int t0  = qt * 16;
    int lane = threadIdx.x;
    const long ld = 3 * C_DIM;

    const bf16_t* qbase = qkv + ((long)(b * T_DIM) + t0) * ld + h * HD;
    v16bf qa0 = load_a_frag(qbase + 0,  ld, lane);
    v16bf qa1 = load_a_frag(qbase + 32, ld, lane);

    v8f o[4] = {v8f{}, v8f{}, v8f{}, v8f{}};
    if (lane < 16) { rowm[lane] = -INFINITY; rowl[lane] = 0.0f; }
    const float sscale = 0.125f;          // 1/sqrt(64)

    for (int kv0 = 0; kv0 <= t0 + 15; kv0 += 32) {
        // ---- S = Q @ K^T over this 32-KV chunk (two 16x16 tiles) ----
#pragma unroll
        for (int st = 0; st < 2; ++st) {
            int n0 = kv0 + st * 16;
            const bf16_t* kbase = qkv + ((long)(b * T_DIM) + n0) * ld + C_DIM + h * HD;
            v8f s = {};
            s = wmma_bf16(qa0, load_bT_frag(kbase + 0,  ld, lane), s);
            s = wmma_bf16(qa1, load_bT_frag(kbase + 32, ld, lane), s);
            int col = lane & 15, rbase = (lane >> 4) ? 8 : 0;
            int kv = n0 + col;
#pragma unroll
            for (int r = 0; r < 8; ++r) {
                int t = t0 + rbase + r;
                S[rbase + r][st * 16 + col] = (kv <= t) ? s[r] * sscale : -INFINITY;
            }
        }
        asm volatile("s_wait_dscnt 0x0" ::: "memory");

        // ---- online softmax per row (lanes 0..15 own one row each) ----
        if (lane < 16) {
            float mold = rowm[lane];
            float mx = mold;
#pragma unroll
            for (int j = 0; j < 32; ++j) mx = fmaxf(mx, S[lane][j]);
            float c = (mold == -INFINITY) ? 0.0f : __expf(mold - mx);
            float sum = 0.0f;
#pragma unroll
            for (int j = 0; j < 32; ++j) {
                float p = __expf(S[lane][j] - mx);
                S[lane][j] = p;
                sum += p;
            }
            rowl[lane] = rowl[lane] * c + sum;
            rowm[lane] = mx;
            rowc[lane] = c;
        }
        asm volatile("s_wait_dscnt 0x0" ::: "memory");

        // ---- rescale O accumulators by per-row correction ----
        {
            int rbase = (lane >> 4) ? 8 : 0;
#pragma unroll
            for (int r = 0; r < 8; ++r) {
                float cc = rowc[rbase + r];
                o[0][r] *= cc; o[1][r] *= cc; o[2][r] *= cc; o[3][r] *= cc;
            }
        }

        // ---- P @ V : A frag (16x32) from LDS, V^T tiles as B frags ----
        v16bf pa;
        {
            int row = lane & 15, kb = (lane >> 4) << 4;
            float4 lo = *(const float4*)&S[row][kb];
            float4 l1 = *(const float4*)&S[row][kb + 4];
            float4 hi = *(const float4*)&S[row][kb + 8];
            float4 h1 = *(const float4*)&S[row][kb + 12];
            pa[0] = f2bf(lo.x); pa[1] = f2bf(lo.y); pa[2] = f2bf(lo.z); pa[3] = f2bf(lo.w);
            pa[4] = f2bf(l1.x); pa[5] = f2bf(l1.y); pa[6] = f2bf(l1.z); pa[7] = f2bf(l1.w);
            pa[8] = f2bf(hi.x); pa[9] = f2bf(hi.y); pa[10] = f2bf(hi.z); pa[11] = f2bf(hi.w);
            pa[12] = f2bf(h1.x); pa[13] = f2bf(h1.y); pa[14] = f2bf(h1.z); pa[15] = f2bf(h1.w);
        }
        const bf16_t* vtb = vT + ((long)(b * H_DIM + h) * HD) * T_DIM;
#pragma unroll
        for (int vt = 0; vt < 4; ++vt) {
            // B[k][n] = V[kv0+k][vt*16+n] = vT[vt*16+n][kv0+k] (contiguous in k)
            const bf16_t* p = vtb + (long)(vt * 16 + (lane & 15)) * T_DIM
                                  + kv0 + ((lane >> 4) << 4);
            v16bf vb = pack16(*(const v8bf*)p, *(const v8bf*)(p + 8));
            o[vt] = wmma_bf16(pa, vb, o[vt]);
        }
        asm volatile("s_wait_dscnt 0x0" ::: "memory");
    }

    // ---- finalize: divide by l, store bf16 into (BT x C) head slice ----
    int col = lane & 15, rbase = (lane >> 4) ? 8 : 0;
#pragma unroll
    for (int r = 0; r < 8; ++r) {
        float inv = 1.0f / rowl[rbase + r];
        long rowg = (long)(b * T_DIM) + t0 + rbase + r;
#pragma unroll
        for (int vt = 0; vt < 4; ++vt)
            outA[rowg * C_DIM + h * HD + vt * 16 + col] = f2bf(o[vt][r] * inv);
    }
}

// ---------------------------------------------------------------------------
// LayerNorm over C per row -> bf16 out (one 256-thread block per row)
// ---------------------------------------------------------------------------
__device__ __forceinline__ float block_reduce_sum256(float v, float* red) {
    red[threadIdx.x] = v; __syncthreads();
    for (int s = 128; s > 0; s >>= 1) {
        if ((int)threadIdx.x < s) red[threadIdx.x] += red[threadIdx.x + s];
        __syncthreads();
    }
    float r = red[0]; __syncthreads();
    return r;
}

__global__ __launch_bounds__(256) void ln_bf16_kernel(
    const float* __restrict__ x, const float* __restrict__ g,
    const float* __restrict__ bta, bf16_t* __restrict__ out)
{
    __shared__ float red[256];
    long row = blockIdx.x;
    const float* xr = x + row * C_DIM;
    float s = 0.0f;
    for (int c = threadIdx.x; c < C_DIM; c += 256) s += xr[c];
    float mu = block_reduce_sum256(s, red) * (1.0f / C_DIM);
    float s2 = 0.0f;
    for (int c = threadIdx.x; c < C_DIM; c += 256) { float d = xr[c] - mu; s2 += d * d; }
    float rstd = rsqrtf(block_reduce_sum256(s2, red) * (1.0f / C_DIM) + 1e-5f);
    for (int c = threadIdx.x; c < C_DIM; c += 256)
        out[row * C_DIM + c] = f2bf((xr[c] - mu) * rstd * g[c] + bta[c]);
}

// mgate path: val = pre + rvec[b]; LN; sigmoid; xcur += sig * retr[b]
__global__ __launch_bounds__(256) void ln_sig_add_kernel(
    const float* __restrict__ pre, const float* __restrict__ rvec,
    const float* __restrict__ retr, const float* __restrict__ g,
    const float* __restrict__ bta, float* __restrict__ xcur)
{
    __shared__ float red[256];
    long row = blockIdx.x;
    int b = (int)(row / T_DIM);
    const float* pr = pre + row * C_DIM;
    const float* rv = rvec + (long)b * C_DIM;
    const float* rt = retr + (long)b * C_DIM;
    float s = 0.0f;
    for (int c = threadIdx.x; c < C_DIM; c += 256) s += pr[c] + rv[c];
    float mu = block_reduce_sum256(s, red) * (1.0f / C_DIM);
    float s2 = 0.0f;
    for (int c = threadIdx.x; c < C_DIM; c += 256) {
        float d = pr[c] + rv[c] - mu; s2 += d * d;
    }
    float rstd = rsqrtf(block_reduce_sum256(s2, red) * (1.0f / C_DIM) + 1e-5f);
    for (int c = threadIdx.x; c < C_DIM; c += 256) {
        float v = (pr[c] + rv[c] - mu) * rstd * g[c] + bta[c];
        float sg = 1.0f / (1.0f + __expf(-v));
        xcur[row * C_DIM + c] += sg * rt[c];
    }
}

// ---------------------------------------------------------------------------
// Small utility kernels
// ---------------------------------------------------------------------------
__global__ void cast_f32_bf16_kernel(const float* __restrict__ src,
                                     bf16_t* __restrict__ dst, long n) {
    long stride = (long)gridDim.x * blockDim.x;
    for (long i = blockIdx.x * (long)blockDim.x + threadIdx.x; i < n; i += stride)
        dst[i] = f2bf(src[i]);
}

// W (K x N) f32 -> Wt (N x K) bf16
__global__ void cast_transpose_kernel(const float* __restrict__ src,
                                      bf16_t* __restrict__ dst, int K, int N) {
    long n = (long)K * N;
    long stride = (long)gridDim.x * blockDim.x;
    for (long i = blockIdx.x * (long)blockDim.x + threadIdx.x; i < n; i += stride) {
        long nn = i / K, kk = i % K;
        dst[i] = f2bf(src[kk * (long)N + nn]);
    }
}

// conv_w (C_out,C_in,3) -> Wt (C x 3C): Wt[co][tap*C+ci] = w[(co*C+ci)*3+tap]
__global__ void repack_convT_kernel(const float* __restrict__ w, bf16_t* __restrict__ dst) {
    long n = 3L * C_DIM * C_DIM;
    long stride = (long)gridDim.x * blockDim.x;
    for (long i = blockIdx.x * (long)blockDim.x + threadIdx.x; i < n; i += stride) {
        long co  = i / (3 * C_DIM);
        long kk  = i % (3 * C_DIM);
        long tap = kk / C_DIM;
        long ci  = kk % C_DIM;
        dst[i] = f2bf(w[(co * C_DIM + ci) * 3 + tap]);
    }
}

// qkv (BT x 3C) -> vT [b][h][d][t]
__global__ void repack_vT_kernel(const bf16_t* __restrict__ qkv, bf16_t* __restrict__ vT) {
    long n = (long)B_DIM * H_DIM * HD * T_DIM;
    long stride = (long)gridDim.x * blockDim.x;
    for (long i = blockIdx.x * (long)blockDim.x + threadIdx.x; i < n; i += stride) {
        long t = i % T_DIM;
        long rest = i / T_DIM;
        long d = rest % HD;
        long h = (rest / HD) % H_DIM;
        long b = rest / ((long)HD * H_DIM);
        vT[i] = qkv[(b * T_DIM + t) * (3 * C_DIM) + 2 * C_DIM + h * HD + d];
    }
}

// mean over T: dst[b*dstStride + off + c] = mean_t src[b,t,c]
__global__ void mean_T_kernel(const float* __restrict__ src, float* __restrict__ dst,
                              int off, int dstStride) {
    int tid = blockIdx.x * blockDim.x + threadIdx.x;
    if (tid >= B_DIM * C_DIM) return;
    int b = tid / C_DIM, c = tid % C_DIM;
    float s = 0.0f;
    const float* p = src + (long)b * T_DIM * C_DIM + c;
    for (int t = 0; t < T_DIM; ++t) s += p[(long)t * C_DIM];
    dst[(long)b * dstStride + off + c] = s * (1.0f / T_DIM);
}

// gate: gw(b,2) = softmax( gelu(LN(gin@g1w+g1b)) @ g2w + g2b )
__global__ __launch_bounds__(256) void gate_kernel(
    const float* __restrict__ gin, const float* __restrict__ g1w,
    const float* __restrict__ g1b, const float* __restrict__ lng,
    const float* __restrict__ lnb, const float* __restrict__ g2w,
    const float* __restrict__ g2b, float* __restrict__ gw)
{
    __shared__ float u[C_DIM];
    __shared__ float red[256];
    int b = blockIdx.x;
    const float* gi = gin + (long)b * 2 * C_DIM;
    for (int j = threadIdx.x; j < C_DIM; j += 256) {
        float s = g1b[j];
        for (int k = 0; k < 2 * C_DIM; ++k) s += gi[k] * g1w[(long)k * C_DIM + j];
        u[j] = s;
    }
    __syncthreads();
    float s = 0.0f;
    for (int j = threadIdx.x; j < C_DIM; j += 256) s += u[j];
    float mu = block_reduce_sum256(s, red) * (1.0f / C_DIM);
    float s2 = 0.0f;
    for (int j = threadIdx.x; j < C_DIM; j += 256) { float d = u[j] - mu; s2 += d * d; }
    float rstd = rsqrtf(block_reduce_sum256(s2, red) * (1.0f / C_DIM) + 1e-5f);
    float p0 = 0.0f, p1 = 0.0f;
    for (int j = threadIdx.x; j < C_DIM; j += 256) {
        float v = gelu_exact((u[j] - mu) * rstd * lng[j] + lnb[j]);
        p0 += v * g2w[j * 2 + 0];
        p1 += v * g2w[j * 2 + 1];
    }
    float l0 = block_reduce_sum256(p0, red);
    float l1 = block_reduce_sum256(p1, red);
    if (threadIdx.x == 0) {
        l0 += g2b[0]; l1 += g2b[1];
        float mx = fmaxf(l0, l1);
        float e0 = __expf(l0 - mx), e1 = __expf(l1 - mx);
        float inv = 1.0f / (e0 + e1);
        gw[b * 2 + 0] = e0 * inv;
        gw[b * 2 + 1] = e1 * inv;
    }
}

// xcur += gw0*fast + gw1*slow
__global__ void combine_kernel(float* __restrict__ x, const float* __restrict__ fast,
                               const float* __restrict__ slow, const float* __restrict__ gw) {
    long n = (long)BT * C_DIM;
    long stride = (long)gridDim.x * blockDim.x;
    for (long i = blockIdx.x * (long)blockDim.x + threadIdx.x; i < n; i += stride) {
        int b = (int)(i / ((long)T_DIM * C_DIM));
        x[i] += gw[b * 2 + 0] * fast[i] + gw[b * 2 + 1] * slow[i];
    }
}

// tiny f32 GEMM: out[r*N+n] = bias[n] + sum_k A[r*K+k]*W[k*N+n]
__global__ void small_linear_kernel(const float* __restrict__ A, const float* __restrict__ W,
                                    const float* __restrict__ bias, float* __restrict__ out,
                                    int rows, int K, int N) {
    int tid = blockIdx.x * blockDim.x + threadIdx.x;
    if (tid >= rows * N) return;
    int r = tid / N, n = tid % N;
    float s = bias ? bias[n] : 0.0f;
    const float* a = A + (long)r * K;
    for (int k = 0; k < K; ++k) s += a[k] * W[(long)k * N + n];
    out[tid] = s;
}

// retr[b] = softmax(mq[b] . mk^T / 32) @ mv      (M_MEM = 16)
__global__ __launch_bounds__(256) void mem_retr_kernel(
    const float* __restrict__ mq, const float* __restrict__ mk,
    const float* __restrict__ mv, float* __restrict__ retr)
{
    __shared__ float w[M_MEM];
    int b = blockIdx.x;
    if (threadIdx.x < M_MEM) {
        float s = 0.0f;
        const float* q = mq + (long)b * C_DIM;
        const float* k = mk + (long)threadIdx.x * C_DIM;
        for (int c = 0; c < C_DIM; ++c) s += q[c] * k[c];
        w[threadIdx.x] = s * (1.0f / 32.0f);   // 1/sqrt(1024)
    }
    __syncthreads();
    if (threadIdx.x == 0) {
        float mx = -INFINITY;
        for (int m = 0; m < M_MEM; ++m) mx = fmaxf(mx, w[m]);
        float sum = 0.0f;
        for (int m = 0; m < M_MEM; ++m) { w[m] = __expf(w[m] - mx); sum += w[m]; }
        float inv = 1.0f / sum;
        for (int m = 0; m < M_MEM; ++m) w[m] *= inv;
    }
    __syncthreads();
    for (int c = threadIdx.x; c < C_DIM; c += 256) {
        float s = 0.0f;
        for (int m = 0; m < M_MEM; ++m) s += w[m] * mv[(long)m * C_DIM + c];
        retr[(long)b * C_DIM + c] = s;
    }
}

// rvec[b][co] = mgate_b[co] + sum_c retr[b][c] * mgate_w[(C+c)*C + co]
__global__ void rvec_kernel(const float* __restrict__ retr, const float* __restrict__ mgw,
                            const float* __restrict__ mgb, float* __restrict__ rvec) {
    int tid = blockIdx.x * blockDim.x + threadIdx.x;
    if (tid >= B_DIM * C_DIM) return;
    int b = tid / C_DIM, co = tid % C_DIM;
    float s = mgb[co];
    const float* r = retr + (long)b * C_DIM;
    for (int c = 0; c < C_DIM; ++c) s += r[c] * mgw[(long)(C_DIM + c) * C_DIM + co];
    rvec[tid] = s;
}

// ---------------------------------------------------------------------------
// Host-side launch
// ---------------------------------------------------------------------------
extern "C" void kernel_launch(void* const* d_in, const int* in_sizes, int n_in,
                              void* d_out, int out_size, void* d_ws, size_t ws_size,
                              hipStream_t stream) {
    const float* x        = (const float*)d_in[0];
    const float* ln1_g    = (const float*)d_in[1];
    const float* ln1_b    = (const float*)d_in[2];
    const float* qkv_w    = (const float*)d_in[3];
    const float* qkv_b    = (const float*)d_in[4];
    const float* aout_w   = (const float*)d_in[5];
    const float* aout_b   = (const float*)d_in[6];
    const float* fln_g    = (const float*)d_in[7];
    const float* fln_b    = (const float*)d_in[8];
    const float* fast_w   = (const float*)d_in[9];
    const float* fast_b   = (const float*)d_in[10];
    const float* sln_g    = (const float*)d_in[11];
    const float* sln_b    = (const float*)d_in[12];
    const float* slow_w   = (const float*)d_in[13];
    const float* slow_b   = (const float*)d_in[14];
    const float* conv_w   = (const float*)d_in[15];
    const float* conv_b   = (const float*)d_in[16];
    const float* g1_w     = (const float*)d_in[17];
    const float* g1_b     = (const float*)d_in[18];
    const float* gln_g    = (const float*)d_in[19];
    const float* gln_b    = (const float*)d_in[20];
    const float* g2_w     = (const float*)d_in[21];
    const float* g2_b     = (const float*)d_in[22];
    const float* mem_bank = (const float*)d_in[23];
    const float* mq_w     = (const float*)d_in[24];
    const float* mq_b     = (const float*)d_in[25];
    const float* mk_w     = (const float*)d_in[26];
    const float* mk_b     = (const float*)d_in[27];
    const float* mv_w     = (const float*)d_in[28];
    const float* mv_b     = (const float*)d_in[29];
    const float* mg_w     = (const float*)d_in[30];
    const float* mg_b     = (const float*)d_in[31];
    const float* mgln_g   = (const float*)d_in[32];
    const float* mgln_b   = (const float*)d_in[33];
    const float* ln2_g    = (const float*)d_in[34];
    const float* ln2_b    = (const float*)d_in[35];
    const float* ff1_w    = (const float*)d_in[36];
    const float* ff1_b    = (const float*)d_in[37];
    const float* ff2_w    = (const float*)d_in[38];
    const float* ff2_b    = (const float*)d_in[39];
    float* out = (float*)d_out;

    // ---- workspace layout ----
    char* ws = (char*)d_ws;
    size_t off = 0;
    auto alloc = [&](size_t bytes) -> void* {
        void* p = ws + off;
        off = (off + bytes + 255) & ~(size_t)255;
        return p;
    };
    bf16_t* wqkvT   = (bf16_t*)alloc((size_t)3 * C_DIM * C_DIM * 2);  // (3C x C)
    bf16_t* wattnT  = (bf16_t*)alloc((size_t)C_DIM * C_DIM * 2);
    bf16_t* wfastT  = (bf16_t*)alloc((size_t)C_DIM * C_DIM * 2);
    bf16_t* wslowT  = (bf16_t*)alloc((size_t)C_DIM * C_DIM * 2);
    bf16_t* wconvT  = (bf16_t*)alloc((size_t)C_DIM * 3 * C_DIM * 2);  // (C x 3C)
    bf16_t* wmgtopT = (bf16_t*)alloc((size_t)C_DIM * C_DIM * 2);
    bf16_t* wff1T   = (bf16_t*)alloc((size_t)4 * C_DIM * C_DIM * 2);  // (4C x C)
    bf16_t* wff2T   = (bf16_t*)alloc((size_t)C_DIM * 4 * C_DIM * 2);  // (C x 4C)
    float*  xcur    = (float*)alloc((size_t)BT * C_DIM * 4);
    bf16_t* abuf    = (bf16_t*)alloc((size_t)BT * C_DIM * 2);
    bf16_t* qkvbf   = (bf16_t*)alloc((size_t)BT * 3 * C_DIM * 2);
    bf16_t* vTbuf   = (bf16_t*)alloc((size_t)BT * C_DIM * 2);
    float*  tmp0    = (float*)alloc((size_t)BT * C_DIM * 4);
    float*  tmp1    = (float*)alloc((size_t)BT * C_DIM * 4);
    bf16_t* ffbf    = (bf16_t*)alloc((size_t)BT * 4 * C_DIM * 2);
    float*  gin     = (float*)alloc((size_t)B_DIM * 2 * C_DIM * 4);
    float*  gwv     = (float*)alloc(64);
    float*  xmean   = (float*)alloc((size_t)B_DIM * C_DIM * 4);
    float*  mq      = (float*)alloc((size_t)B_DIM * C_DIM * 4);
    float*  mk      = (float*)alloc((size_t)M_MEM * C_DIM * 4);
    float*  mv      = (float*)alloc((size_t)M_MEM * C_DIM * 4);
    float*  retr    = (float*)alloc((size_t)B_DIM * C_DIM * 4);
    float*  rvec    = (float*)alloc((size_t)B_DIM * C_DIM * 4);

    auto castT = [&](const float* s, bf16_t* d, int K, int N) {
        cast_transpose_kernel<<<1024, 256, 0, stream>>>(s, d, K, N);
    };
    auto cast = [&](const float* s, bf16_t* d, long n) {
        cast_f32_bf16_kernel<<<1024, 256, 0, stream>>>(s, d, n);
    };
    // B transposed (Wt: N x K). grid covers (M/256)*(N/64) blocks exactly.
    auto gemm = [&](const bf16_t* A, int lda, const bf16_t* Wt, int ldk,
                    const float* bias, const float* res, float* oF, bf16_t* oB,
                    int M, int N, int K, int act, float scale) {
        int blocks = (M / 256) * (N / 64);
        gemm_bf16_kernel<<<blocks, 256, 0, stream>>>(A, lda, Wt, ldk, bias, res,
                                                     oF, oB, M, N, K, act, scale);
    };

    // ---- 0) weights -> transposed bf16 ----
    castT(qkv_w, wqkvT, C_DIM, 3 * C_DIM);
    castT(aout_w, wattnT, C_DIM, C_DIM);
    castT(fast_w, wfastT, C_DIM, C_DIM);
    castT(slow_w, wslowT, C_DIM, C_DIM);
    castT(mg_w, wmgtopT, C_DIM, C_DIM);          // top C rows of (2C x C)
    castT(ff1_w, wff1T, C_DIM, 4 * C_DIM);
    castT(ff2_w, wff2T, 4 * C_DIM, C_DIM);
    repack_convT_kernel<<<1024, 256, 0, stream>>>(conv_w, wconvT);

    // ---- 1) attention block ----
    ln_bf16_kernel<<<BT, 256, 0, stream>>>(x, ln1_g, ln1_b, abuf);
    gemm(abuf, C_DIM, wqkvT, C_DIM, qkv_b, nullptr, nullptr, qkvbf,
         BT, 3 * C_DIM, C_DIM, 0, 1.0f);
    repack_vT_kernel<<<1024, 256, 0, stream>>>(qkvbf, vTbuf);
    attn_kernel<<<B_DIM * H_DIM * (T_DIM / 16), 32, 0, stream>>>(qkvbf, vTbuf, abuf);
    gemm(abuf, C_DIM, wattnT, C_DIM, aout_b, x, xcur, nullptr,
         BT, C_DIM, C_DIM, 0, 1.0f);

    // ---- 2) dual pathway ----
    ln_bf16_kernel<<<BT, 256, 0, stream>>>(xcur, fln_g, fln_b, abuf);
    gemm(abuf, C_DIM, wfastT, C_DIM, fast_b, nullptr, tmp0, nullptr,
         BT, C_DIM, C_DIM, 1, 1.0f);                     // gelu -> fast
    ln_bf16_kernel<<<BT, 256, 0, stream>>>(xcur, sln_g, sln_b, abuf);
    gemm(abuf, C_DIM, wslowT, C_DIM, slow_b, nullptr, tmp1, nullptr,
         BT, C_DIM, C_DIM, 2, 1.0f);                     // tanh -> slow_base
    cast(xcur, abuf, (long)BT * C_DIM);
    conv_gemm_kernel<<<(BT / 256) * (C_DIM / 64), 256, 0, stream>>>(
        abuf, wconvT, conv_b, tmp1);
    mean_T_kernel<<<(B_DIM * C_DIM + 255) / 256, 256, 0, stream>>>(tmp0, gin, 0, 2 * C_DIM);
    mean_T_kernel<<<(B_DIM * C_DIM + 255) / 256, 256, 0, stream>>>(tmp1, gin, C_DIM, 2 * C_DIM);
    gate_kernel<<<B_DIM, 256, 0, stream>>>(gin, g1_w, g1_b, gln_g, gln_b, g2_w, g2_b, gwv);
    combine_kernel<<<1024, 256, 0, stream>>>(xcur, tmp0, tmp1, gwv);

    // ---- 3) memory retrieval ----
    mean_T_kernel<<<(B_DIM * C_DIM + 255) / 256, 256, 0, stream>>>(xcur, xmean, 0, C_DIM);
    small_linear_kernel<<<(B_DIM * C_DIM + 255) / 256, 256, 0, stream>>>(
        xmean, mq_w, mq_b, mq, B_DIM, C_DIM, C_DIM);
    small_linear_kernel<<<(M_MEM * C_DIM + 255) / 256, 256, 0, stream>>>(
        mem_bank, mk_w, mk_b, mk, M_MEM, C_DIM, C_DIM);
    small_linear_kernel<<<(M_MEM * C_DIM + 255) / 256, 256, 0, stream>>>(
        mem_bank, mv_w, mv_b, mv, M_MEM, C_DIM, C_DIM);
    mem_retr_kernel<<<B_DIM, 256, 0, stream>>>(mq, mk, mv, retr);
    rvec_kernel<<<(B_DIM * C_DIM + 255) / 256, 256, 0, stream>>>(retr, mg_w, mg_b, rvec);
    cast(xcur, abuf, (long)BT * C_DIM);
    gemm(abuf, C_DIM, wmgtopT, C_DIM, nullptr, nullptr, tmp0, nullptr,
         BT, C_DIM, C_DIM, 0, 1.0f);                     // pre-LN mgate term
    ln_sig_add_kernel<<<BT, 256, 0, stream>>>(tmp0, rvec, retr, mgln_g, mgln_b, xcur);

    // ---- 4) feed-forward ----
    ln_bf16_kernel<<<BT, 256, 0, stream>>>(xcur, ln2_g, ln2_b, abuf);
    gemm(abuf, C_DIM, wff1T, C_DIM, ff1_b, nullptr, nullptr, ffbf,
         BT, 4 * C_DIM, C_DIM, 1, 1.0f);                 // gelu -> bf16
    gemm(ffbf, 4 * C_DIM, wff2T, 4 * C_DIM, ff2_b, xcur, out, nullptr,
         BT, C_DIM, 4 * C_DIM, 0, 1.0f);                 // final residual -> d_out
}
